// CharacterTokenEmbedder_26843545600318
// MI455X (gfx1250) — matrix operand order repaired and checked by reading, not compiled, not run on hardware
//
#include <hip/hip_runtime.h>
#include <stdint.h>

typedef __attribute__((ext_vector_type(16))) _Float16 v16h;
typedef __attribute__((ext_vector_type(8)))  float    v8f;
typedef __attribute__((ext_vector_type(4)))  unsigned int u32x4;
typedef __attribute__((ext_vector_type(8)))  int      i32x8;
typedef __attribute__((ext_vector_type(4)))  int      i32x4;

#define DEV __device__ __forceinline__

// ---------------------------------------------------------------------------
// WMMA fragment helpers (CDNA5 16x16x32 f16, wave32 layouts per ISA 7.12.2)
// ---------------------------------------------------------------------------
DEV v16h frag_a(const _Float16* p, int ldh) {
  int lane = threadIdx.x & 31;
  int m = lane & 15, g = lane >> 4;
  const _Float16* row = p + (size_t)m * ldh + 8 * g;
  v16h out;
  ((float4*)&out)[0] = *(const float4*)(row);
  ((float4*)&out)[1] = *(const float4*)(row + 16);
  return out;
}

DEV v16h frag_b(const _Float16* p, int ldh) {
  int lane = threadIdx.x & 31;
  int n = lane & 15, g = lane >> 4;
  const _Float16* row = p + (size_t)n * ldh + 8 * g;
  v16h out;
  ((float4*)&out)[0] = *(const float4*)(row);
  ((float4*)&out)[1] = *(const float4*)(row + 16);
  return out;
}

DEV v8f wmma_f16(v16h a, v16h b, v8f c) {
  return __builtin_amdgcn_wmma_f32_16x16x32_f16(
      false, a, false, b, (short)0, c, false, false);
}

// ---------------------------------------------------------------------------
// TDM: Tensor Data Mover 2D tile load Global -> LDS (ISA ch.8 D# layout).
// One instruction DMAs a tile_dim0 x tile_dim1 tile (2-byte elements) of a
// row-major tensor [tdim1][tdim0] with row stride `stride0` into LDS.
// Tracked with TENSORcnt.  Issue from a single wave; others sync via barrier.
// 6-arg builtin form (clang-23 / therock-10.0 toolchain).
// ---------------------------------------------------------------------------
DEV void tdm_load_2d_f16(uint32_t lds_off, const void* gaddr,
                         uint32_t tdim0, uint32_t tdim1,
                         uint32_t tile0, uint32_t tile1, uint64_t stride0) {
  uint64_t ga = (uint64_t)(uintptr_t)gaddr;
  u32x4 g0;
  g0.x = 1u;                                                // count=1
  g0.y = lds_off;                                           // lds_addr (bytes)
  g0.z = (uint32_t)ga;                                      // global_addr[31:0]
  g0.w = (uint32_t)((ga >> 32) & 0x01FFFFFFu) | (2u << 30); // addr[56:32]|type=2
  i32x8 g1;
  g1[0] = (int)(1u << 16);                                  // data_size=1 (2B)
  g1[1] = (int)((tdim0 & 0xFFFFu) << 16);                   // tensor_dim0 lo
  g1[2] = (int)((tdim0 >> 16) | ((tdim1 & 0xFFFFu) << 16)); // dim0 hi | dim1 lo
  g1[3] = (int)((tdim1 >> 16) | (tile0 << 16));             // dim1 hi | tile0
  g1[4] = (int)(tile1 & 0xFFFFu);                           // tile1, tile2=0
  g1[5] = (int)(uint32_t)(stride0 & 0xFFFFFFFFu);           // dim0_stride lo
  g1[6] = (int)(uint32_t)((stride0 >> 32) & 0xFFFFu);       // dim0_stride hi
  g1[7] = 0;
  i32x4 z4 = {0, 0, 0, 0};
  i32x8 z8 = {0, 0, 0, 0, 0, 0, 0, 0};
  __builtin_amdgcn_tensor_load_to_lds(g0, g1, z4, z4, z8, 0);
}

DEV void wait_tensor0() { __builtin_amdgcn_s_wait_tensorcnt(0); }

// Per-lane async global->LDS b128 copy, tracked with ASYNCcnt.
DEV void async_cp16(uint32_t lds_off, const void* g) {
  asm volatile("global_load_async_to_lds_b128 %0, %1, off"
               :: "v"(lds_off), "v"((uint64_t)(uintptr_t)g)
               : "memory");
}
DEV void wait_async0() { asm volatile("s_wait_asynccnt 0x0" ::: "memory"); }

// ---------------------------------------------------------------------------
// Weight prep kernels
// ---------------------------------------------------------------------------
__global__ void k_cvt_f32_f16(const float* __restrict__ src,
                              uint16_t* __restrict__ dst, int n) {
  int i = blockIdx.x * blockDim.x + threadIdx.x;
  if (i < n) ((_Float16*)dst)[i] = (_Float16)src[i];
}

__global__ void k_pack_conv_w(const float* __restrict__ w,
                              uint16_t* __restrict__ dst,
                              int oc, int width, int Kp) {
  int i = blockIdx.x * blockDim.x + threadIdx.x;
  if (i >= oc * Kp) return;
  int o = i / Kp, j = i - o * Kp;
  int k = j >> 4, c = j & 15;
  float v = (k < width) ? w[(o * 16 + c) * width + k] : 0.0f;
  ((_Float16*)dst)[i] = (_Float16)v;
}

__global__ void k_pack_bias(const float* b0, const float* b1, const float* b2,
                            const float* b3, const float* b4, const float* b5,
                            const float* b6, float* __restrict__ dst) {
  int i = blockIdx.x * blockDim.x + threadIdx.x;
  if (i >= 2048) return;
  const float* bs[7] = {b0, b1, b2, b3, b4, b5, b6};
  const int base[8] = {0, 32, 64, 128, 256, 512, 1024, 2048};
  int f = 0;
  while (i >= base[f + 1]) ++f;
  dst[i] = bs[f][i - base[f]];
}

// ---------------------------------------------------------------------------
// Stage 1: char gather + embedding -> Xc [N][64][16] f16 (rows 50..63 zero)
// ---------------------------------------------------------------------------
__global__ void k_embed(const int* __restrict__ words, const int* __restrict__ wtc,
                        const float* __restrict__ cemb, uint16_t* __restrict__ Xc) {
  __shared__ int sCid[64];
  int t = blockIdx.x, tid = threadIdx.x;
  int word = words[t];
  if (tid < 64) sCid[tid] = (tid < 50) ? wtc[(size_t)word * 50 + tid] : 0;
  __syncthreads();
  _Float16* X = (_Float16*)Xc + (size_t)t * 1024;
  for (int i = tid; i < 1024; i += blockDim.x) {
    int r = i >> 4, c = i & 15;
    float v = (r < 50) ? cemb[sCid[r] * 16 + c] : 0.0f;
    X[i] = (_Float16)v;
  }
}

// ---------------------------------------------------------------------------
// Stage 2: convs via WMMA (free im2col), masked max-over-time, relu
// ---------------------------------------------------------------------------
__global__ void __launch_bounds__(256)
k_conv(const uint16_t* __restrict__ Xc, const uint16_t* __restrict__ Wc,
       const float* __restrict__ bconv, uint16_t* __restrict__ h) {
  static const int kWdt[7]  = {1, 2, 3, 4, 5, 6, 7};
  static const int kKp[7]   = {32, 32, 64, 64, 96, 96, 128};
  static const int kMt[7]   = {4, 4, 3, 3, 3, 3, 3};
  static const int kTile[8] = {0, 2, 4, 8, 16, 32, 64, 128};
  static const int kCol[7]  = {0, 32, 64, 128, 256, 512, 1024};
  static const int kWOff[7] = {0, 1024, 2048, 6144, 14336, 38912, 88064};

  __shared__ __align__(16) _Float16 sX[1152];
  int t = blockIdx.x, tid = threadIdx.x;
  int lane = tid & 31, wave = tid >> 5;

  {
    uint32_t* s32 = (uint32_t*)sX;
    const uint32_t* g32 = (const uint32_t*)(Xc + (size_t)t * 1024);
    s32[tid] = g32[tid];
    s32[tid + 256] = g32[tid + 256];
    if (tid < 64) s32[512 + tid] = 0u;
  }
  __syncthreads();

  const _Float16* W16 = (const _Float16*)Wc;
  int g = lane >> 4;
  for (int ot = wave; ot < 128; ot += 8) {
    int f = 0;
    while (ot >= kTile[f + 1]) ++f;
    int lt = ot - kTile[f];
    int Kp = kKp[f], Ks = Kp >> 5;
    int Pv = 51 - kWdt[f];
    float cmax = -1e30f;
    for (int mt = 0; mt < kMt[f]; ++mt) {
      v8f acc = {};
      const _Float16* Abase = sX + mt * 256;
      const _Float16* Bbase = W16 + kWOff[f] + (size_t)(lt * 16) * Kp;
      for (int ks = 0; ks < Ks; ++ks) {
        v16h a = frag_a(Abase + ks * 32, 16);
        v16h b = frag_b(Bbase + ks * 32, Kp);
        acc = wmma_f16(a, b, acc);
      }
#pragma unroll
      for (int v = 0; v < 8; ++v) {
        int row = mt * 16 + v + 8 * g;
        float x = (row < Pv) ? acc[v] : -1e30f;
        cmax = fmaxf(cmax, x);
      }
    }
    cmax = fmaxf(cmax, __shfl_xor(cmax, 16, 32));
    int col = kCol[f] + lt * 16 + (lane & 15);
    float y = fmaxf(cmax + bconv[col], 0.0f);
    if (lane < 16) ((_Float16*)h)[(size_t)t * 2048 + col] = (_Float16)y;
  }
}

// ---------------------------------------------------------------------------
// Stage 3: fused highway layer, M_BLK=64, TDM double-buffered A staging.
// Each wave: one 16-col tile, proj+gate, 4 M subtiles -> 8 WMMA per K-step.
// ---------------------------------------------------------------------------
__global__ void __launch_bounds__(256)
k_highway(const uint16_t* __restrict__ hin_, const uint16_t* __restrict__ W_,
          const float* __restrict__ bias, uint16_t* __restrict__ hout_) {
  __shared__ __align__(16) _Float16 sA[2][64 * 32];
  const _Float16* hin = (const _Float16*)hin_;
  const _Float16* W = (const _Float16*)W_;
  _Float16* hout = (_Float16*)hout_;
  int tid = threadIdx.x, lane = tid & 31, wave = tid >> 5;
  int m0 = blockIdx.x * 64;
  int jc = (blockIdx.y * 8 + wave) * 16;
  uint32_t ldsOff[2] = {(uint32_t)(uintptr_t)&sA[0][0],
                        (uint32_t)(uintptr_t)&sA[1][0]};
  const _Float16* gA = hin + (size_t)m0 * 2048;  // 64x2048 sub-panel

  v8f accP[4] = {}, accG[4] = {};

  // prologue: DMA K-step 0 tile (64 rows x 32 cols) into buffer 0
  if (wave == 0) {
    tdm_load_2d_f16(ldsOff[0], gA, 2048, 4096, 32, 64, 2048);
    wait_tensor0();
  }
  __syncthreads();

  for (int ks = 0; ks < 64; ++ks) {
    int cur = ks & 1;
    if (wave == 0 && ks + 1 < 64)  // prefetch next K tile into other buffer
      tdm_load_2d_f16(ldsOff[cur ^ 1], gA + (ks + 1) * 32,
                      2048, 4096, 32, 64, 2048);
    if (ks + 1 < 64) {
      __builtin_prefetch(W + (size_t)jc * 2048 + (ks + 1) * 32, 0, 3);
      __builtin_prefetch(W + (size_t)(jc + 2048) * 2048 + (ks + 1) * 32, 0, 3);
    }
    v16h bp = frag_b(W + (size_t)jc * 2048 + ks * 32, 2048);
    v16h bg = frag_b(W + (size_t)(jc + 2048) * 2048 + ks * 32, 2048);
    const _Float16* buf = &sA[cur][0];
#pragma unroll
    for (int s = 0; s < 4; ++s) {
      v16h a = frag_a(buf + s * 16 * 32, 32);
      accP[s] = wmma_f16(a, bp, accP[s]);
      accG[s] = wmma_f16(a, bg, accG[s]);
    }
    if (wave == 0) wait_tensor0();  // next buffer ready
    __syncthreads();                // everyone done reading cur
  }

  int gL = lane >> 4, col = jc + (lane & 15);
  float bp_ = bias[col], bg_ = bias[col + 2048];
#pragma unroll
  for (int s = 0; s < 4; ++s) {
#pragma unroll
    for (int v = 0; v < 8; ++v) {
      int m = m0 + s * 16 + v + 8 * gL;
      float p = accP[s][v] + bp_;
      float gt = accG[s][v] + bg_;
      float sg = 1.0f / (1.0f + __expf(-gt));
      float hold = (float)hin[(size_t)m * 2048 + col];
      float hn = sg * hold + (1.0f - sg) * fmaxf(p, 0.0f);
      hout[(size_t)m * 2048 + col] = (_Float16)hn;
    }
  }
}

// ---------------------------------------------------------------------------
// Stage 4: projection, M_BLK=64, per-lane async global->LDS A staging,
// +bias, PAD/EOS/UNK overrides -> out [N][512] f32
// ---------------------------------------------------------------------------
__global__ void __launch_bounds__(256)
k_proj(const uint16_t* __restrict__ hin_, const uint16_t* __restrict__ Pw_,
       const float* __restrict__ pb, const int* __restrict__ words,
       const float* __restrict__ symb, float* __restrict__ out) {
  __shared__ __align__(16) _Float16 sA[64 * 32];
  const _Float16* hin = (const _Float16*)hin_;
  const _Float16* Pw = (const _Float16*)Pw_;
  int tid = threadIdx.x, lane = tid & 31, wave = tid >> 5;
  int m0 = blockIdx.x * 64;
  int jc = (blockIdx.y * 8 + wave) * 16;
  int r = tid >> 2, ch = tid & 3;  // 64 rows x four 8-half chunks
  uint32_t myLds = (uint32_t)(uintptr_t)(sA + r * 32 + ch * 8);
  const _Float16* myG = hin + (size_t)(m0 + r) * 2048 + ch * 8;

  v8f acc[4] = {};
  for (int ks = 0; ks < 64; ++ks) {
    __syncthreads();                    // prior reads of sA complete
    async_cp16(myLds, myG + ks * 32);   // ASYNCcnt-tracked b128 to LDS
    wait_async0();
    __syncthreads();
    v16h b = frag_b(Pw + (size_t)jc * 2048 + ks * 32, 2048);
#pragma unroll
    for (int s = 0; s < 4; ++s) {
      v16h a = frag_a(sA + s * 16 * 32, 32);
      acc[s] = wmma_f16(a, b, acc[s]);
    }
  }
  int gL = lane >> 4, col = jc + (lane & 15);
  float bb = pb[col];
#pragma unroll
  for (int s = 0; s < 4; ++s) {
#pragma unroll
    for (int v = 0; v < 8; ++v) {
      int m = m0 + s * 16 + v + 8 * gL;
      float o = acc[s][v] + bb;
      int wid = words[m];
      if (wid == 1) o = 0.0f;                  // PAD
      else if (wid == 2) o = symb[col];        // EOS
      else if (wid == 3) o = symb[512 + col];  // UNK
      out[(size_t)m * 512 + col] = o;
    }
  }
}

// ---------------------------------------------------------------------------
extern "C" void kernel_launch(void* const* d_in, const int* in_sizes, int n_in,
                              void* d_out, int out_size, void* d_ws, size_t ws_size,
                              hipStream_t stream) {
  (void)n_in; (void)out_size; (void)ws_size;
  const int* words = (const int*)d_in[0];
  const int* wtc   = (const int*)d_in[1];
  const float* cemb = (const float*)d_in[2];
  const float* symb = (const float*)d_in[3];
  const float* convw[7]; for (int i = 0; i < 7; ++i) convw[i] = (const float*)d_in[4 + i];
  const float* convb[7]; for (int i = 0; i < 7; ++i) convb[i] = (const float*)d_in[11 + i];
  const float* hww0 = (const float*)d_in[18];
  const float* hww1 = (const float*)d_in[19];
  const float* hwb0 = (const float*)d_in[20];
  const float* hwb1 = (const float*)d_in[21];
  const float* pw = (const float*)d_in[22];
  const float* pb = (const float*)d_in[23];
  float* out = (float*)d_out;

  const int N = in_sizes[0];  // 4096 tokens
  const int D = 2048, HD = 4096, OD = 512;

  size_t off = 0;
  auto carve = [&](size_t bytes) -> void* {
    off = (off + 255) & ~(size_t)255;
    void* p = (char*)d_ws + off;
    off += bytes;
    return p;
  };
  uint16_t* Xc = (uint16_t*)carve((size_t)N * 1024 * 2);
  uint16_t* h0 = (uint16_t*)carve((size_t)N * D * 2);
  uint16_t* h1 = (uint16_t*)carve((size_t)N * D * 2);
  uint16_t* W0 = (uint16_t*)carve((size_t)HD * D * 2);
  uint16_t* W1 = (uint16_t*)carve((size_t)HD * D * 2);
  uint16_t* Pw = (uint16_t*)carve((size_t)OD * D * 2);
  uint16_t* Wc = (uint16_t*)carve((size_t)219136 * 2);
  float*   bcv = (float*)carve(2048 * 4);

  {
    int n = HD * D;
    k_cvt_f32_f16<<<(n + 255) / 256, 256, 0, stream>>>(hww0, W0, n);
    k_cvt_f32_f16<<<(n + 255) / 256, 256, 0, stream>>>(hww1, W1, n);
    int np = OD * D;
    k_cvt_f32_f16<<<(np + 255) / 256, 256, 0, stream>>>(pw, Pw, np);
  }
  {
    static const int oc[7] = {32, 32, 64, 128, 256, 512, 1024};
    static const int wd[7] = {1, 2, 3, 4, 5, 6, 7};
    static const int kp[7] = {32, 32, 64, 64, 96, 96, 128};
    static const int wo[7] = {0, 1024, 2048, 6144, 14336, 38912, 88064};
    for (int f = 0; f < 7; ++f) {
      int n = oc[f] * kp[f];
      k_pack_conv_w<<<(n + 255) / 256, 256, 0, stream>>>(convw[f], Wc + wo[f],
                                                         oc[f], wd[f], kp[f]);
    }
    k_pack_bias<<<8, 256, 0, stream>>>(convb[0], convb[1], convb[2], convb[3],
                                       convb[4], convb[5], convb[6], bcv);
  }
  k_embed<<<N, 256, 0, stream>>>(words, wtc, cemb, Xc);
  k_conv<<<N, 256, 0, stream>>>(Xc, Wc, bcv, h0);
  dim3 ghw(N / 64, 16);
  k_highway<<<ghw, 256, 0, stream>>>(h0, W0, hwb0, h1);
  k_highway<<<ghw, 256, 0, stream>>>(h1, W1, hwb1, h0);
  dim3 gpj(N / 64, 4);
  k_proj<<<gpj, 256, 0, stream>>>(h0, Pw, pb, words, symb, out);
}
